// FullDPM_24283745091644
// MI455X (gfx1250) — matrix operand order repaired and checked by reading, not compiled, not run on hardware
//
#include <hip/hip_runtime.h>
#include <hip/hip_bf16.h>
#include <utility>

// ---------------------------------------------------------------------------
// FullDPM forward for MI455X (gfx1250, wave32, WMMA).
//
// All matmuls use v_wmma_f32_16x16x32_f16 (f16 A/B, f32 accum).
// - Weights pre-transposed/padded to f16: B fragment = 2 contiguous b128
//   loads per lane per k-tile.
// - Concat GEMMs use compile-time staged k-loops (all boundaries are
//   k-tile/chunk aligned) -> branch-free inner loops.
// - NT=2 n-blocking: one A fragment feeds two B tiles / two WMMAs.
// - M always a multiple of 16 -> no row predication.
// ---------------------------------------------------------------------------

#define B_    32
#define L_    256
#define C_    4
#define HID   128
#define KCLS  20
#define KNN   9
#define N_    (B_ * L_)      // 8192  (= 512*16)
#define EC_   (N_ * KNN)     // 73728 ctx edges
#define E_    (2 * EC_)      // 147456 (= 9216*16)
#define BIGF  1e10f
#define STDF  10.0f
#define ILK   160            // padded K for input_linear (131 -> 160)

typedef _Float16 v16h __attribute__((ext_vector_type(16)));
typedef _Float16 v8h  __attribute__((ext_vector_type(8)));
typedef float    v8f  __attribute__((ext_vector_type(8)));

union FragA { v16h v; v8h h8[2]; _Float16 e[16]; };
union FragC { v8f  v; float e[8]; };

__device__ inline v8h zero8() {
  v8h z;
#pragma unroll
  for (int i = 0; i < 8; ++i) z[i] = (_Float16)0.f;
  return z;
}

__device__ inline v8h cvt8(const float* s) {
  const float4 f0 = *(const float4*)(s);
  const float4 f1 = *(const float4*)(s + 4);
  v8h r;
  r[0] = (_Float16)f0.x; r[1] = (_Float16)f0.y; r[2] = (_Float16)f0.z; r[3] = (_Float16)f0.w;
  r[4] = (_Float16)f1.x; r[5] = (_Float16)f1.y; r[6] = (_Float16)f1.z; r[7] = (_Float16)f1.w;
  return r;
}

// ---------------------------------------------------------------------------
// Diffusion schedule (cosine), computed on device (betas[0]=0, len 101).
// ---------------------------------------------------------------------------
__global__ void sched_k(float* betas, float* alphas, float* abar) {
  if (threadIdx.x != 0 || blockIdx.x != 0) return;
  const float s = 0.01f;
  const float PI_2 = 1.57079632679489662f;
  float f0 = cosf(PI_2 * (s / (1.f + s))); f0 *= f0;
  float prev_ab = 1.f;
  for (int t = 0; t <= 100; ++t) {
    float ft = cosf(PI_2 * ((t / 100.f) + s) / (1.f + s)); ft *= ft;
    float ab = ft / f0;
    float beta = 0.f;
    if (t > 0) beta = fminf(fmaxf(1.f - ab / prev_ab, 0.f), 0.999f);
    betas[t] = beta; alphas[t] = 1.f - beta; abar[t] = ab;
    prev_ab = ab;
  }
}

// ---------------------------------------------------------------------------
// Weight prep: W[K,Nreal] f32 -> Wt[Npad][Kpad] f16 (transposed, padded),
// bias -> biasp[Npad] f32 (padded; bias may be null).
// ---------------------------------------------------------------------------
__global__ void wprep_k(const float* W, const float* bias, _Float16* Wt, float* biasp,
                        int K, int Nreal, int Kpad, int Npad) {
  int idx = blockIdx.x * 256 + threadIdx.x;
  int total = Npad * Kpad;
  if (idx < total) {
    int n = idx / Kpad, k = idx % Kpad;
    float v = (k < K && n < Nreal) ? W[(size_t)k * Nreal + n] : 0.f;
    Wt[(size_t)n * Kpad + k] = (_Float16)v;
  }
  if (idx < Npad) biasp[idx] = (bias != nullptr && idx < Nreal) ? bias[idx] : 0.f;
}

// ---------------------------------------------------------------------------
// Per-batch centers from boundary residues.
// ---------------------------------------------------------------------------
__global__ __launch_bounds__(256) void center_k(const float* X0, const unsigned char* amask,
                                                const int* seg, float* centers) {
  int b = blockIdx.x, l = threadIdx.x, g = (b << 8) + l;
  bool mg = (seg[g] == 1);
  bool bnd = (g < N_ - 1) && (mg != (seg[g + 1] == 1));
  float sx = 0, sy = 0, sz = 0, sw = 0;
  if (bnd) {
    for (int c = 0; c < C_; ++c) {
      if (amask[(size_t)g * C_ + c]) {
        sx += X0[(size_t)g * 12 + c * 3 + 0];
        sy += X0[(size_t)g * 12 + c * 3 + 1];
        sz += X0[(size_t)g * 12 + c * 3 + 2];
        sw += 1.f;
      }
    }
  }
  __shared__ float rx[256], ry[256], rz[256], rw[256];
  rx[l] = sx; ry[l] = sy; rz[l] = sz; rw[l] = sw;
  __syncthreads();
  for (int s2 = 128; s2 > 0; s2 >>= 1) {
    if (l < s2) { rx[l] += rx[l+s2]; ry[l] += ry[l+s2]; rz[l] += rz[l+s2]; rw[l] += rw[l+s2]; }
    __syncthreads();
  }
  if (l == 0) {
    float dd = fmaxf(rw[0], 1.f);
    centers[b * 4 + 0] = rx[0] / dd;
    centers[b * 4 + 1] = ry[0] / dd;
    centers[b * 4 + 2] = rz[0] / dd;
    centers[b * 4 + 3] = dd;
  }
}

// ---------------------------------------------------------------------------
// Noising: Xn, p_noisy, gumbel-max s_noisy, t-embedding.
// ---------------------------------------------------------------------------
__global__ __launch_bounds__(256) void noise_k(
    const float* X0, const float* eps, const float* gum,
    const int* S0, const int* seg, const int* tB,
    const int* dsflag, const int* dqflag,
    const float* betas, const float* abar, const float* centers,
    float* pn, float* x0, float* temb, int* snz)
{
  int g = blockIdx.x * 256 + threadIdx.x;
  if (g >= N_) return;
  int b = g >> 8;
  int tb = tB[b];
  float ab = abar[tb];
  float sab = sqrtf(ab), s1ab = sqrtf(fmaxf(1.f - ab, 0.f));
  bool mg = (seg[g] == 1);
  int ds = *dsflag, dq = *dqflag;
  for (int j = 0; j < 12; ++j) {
    float xn = (X0[(size_t)g * 12 + j] - centers[b * 4 + (j % 3)]) * (1.f / STDF);
    float p = xn;
    if (ds && mg) p = sab * xn + s1ab * eps[(size_t)g * 12 + j];
    pn[(size_t)g * 12 + j] = p;
    x0[(size_t)g * 12 + j] = p;
  }
  int s0v = S0[g];
  int sn = s0v;
  if (dq) {
    float best = -1e30f; int bi = 0;
    for (int k = 0; k < KCLS; ++k) {
      float c0  = (k == s0v) ? 1.f : 0.f;
      float ctk = mg ? (ab * c0 + (1.f - ab) / KCLS) : c0;
      float u   = gum[(size_t)g * KCLS + k];
      float gk  = -logf(-logf(u + 1e-12f) + 1e-12f);
      float v   = logf(ctk + 1e-30f) + gk;
      if (v > best) { best = v; bi = k; }
    }
    sn = bi;
  }
  snz[g] = sn;
  float bt = betas[tb];
  temb[g * 4 + 0] = bt;
  temb[g * 4 + 1] = sinf(bt);
  temb[g * 4 + 2] = cosf(bt);
  temb[g * 4 + 3] = 0.f;
}

// ---------------------------------------------------------------------------
// kNN edges: per (b,i) block; min-atom-pair distance, segment-masked,
// exact top-9 with ascending-index tie-break (== jax.lax.top_k order).
// ---------------------------------------------------------------------------
__global__ __launch_bounds__(256) void edges_k(const float* pn, const unsigned char* amask,
                                               const int* seg, int* ctxc, int* itrc) {
  int gi = blockIdx.x;
  int b = gi >> 8;
  int j = threadIdx.x;
  int gj = (b << 8) + j;
  __shared__ float Ai[12];
  __shared__ int Aim, segi;
  __shared__ float dc[256], di[256];
  __shared__ float rv[256];
  __shared__ int   ri[256];
  if (j < 12) Ai[j] = pn[(size_t)gi * 12 + j];
  if (j == 0) {
    int m = 0;
    for (int c = 0; c < C_; ++c) if (amask[(size_t)gi * C_ + c]) m |= 1 << c;
    Aim = m; segi = seg[gi];
  }
  __syncthreads();
  float Aj[12];
  for (int q = 0; q < 12; ++q) Aj[q] = pn[(size_t)gj * 12 + q];
  int mj = 0;
  for (int c = 0; c < C_; ++c) if (amask[(size_t)gj * C_ + c]) mj |= 1 << c;
  float best = BIGF;
  for (int c = 0; c < C_; ++c) {
    if (!((Aim >> c) & 1)) continue;
    for (int d = 0; d < C_; ++d) {
      if (!((mj >> d) & 1)) continue;
      float dx = Ai[c*3+0] - Aj[d*3+0];
      float dy = Ai[c*3+1] - Aj[d*3+1];
      float dz = Ai[c*3+2] - Aj[d*3+2];
      float dist = sqrtf(fmaxf(dx*dx + dy*dy + dz*dz, 0.f));
      best = fminf(best, dist);
    }
  }
  bool same = (segi == seg[gj]);
  dc[j] = same ? best : BIGF;
  di[j] = same ? BIGF : best;
  __syncthreads();
  for (int pass = 0; pass < 2; ++pass) {
    float* dd   = pass ? di   : dc;
    int*   outp = pass ? itrc : ctxc;
    for (int kk = 0; kk < KNN; ++kk) {
      rv[j] = dd[j]; ri[j] = j;
      __syncthreads();
      for (int s2 = 128; s2 > 0; s2 >>= 1) {
        if (j < s2) {
          if (rv[j+s2] < rv[j] || (rv[j+s2] == rv[j] && ri[j+s2] < ri[j])) {
            rv[j] = rv[j+s2]; ri[j] = ri[j+s2];
          }
        }
        __syncthreads();
      }
      if (j == 0) { outp[(size_t)gi * KNN + kk] = (b << 8) + ri[0]; dd[ri[0]] = 2.f * BIGF; }
      __syncthreads();
    }
  }
}

// ---------------------------------------------------------------------------
// h0ext (f16 [N, 160]): [seq_emb[s]+pos_emb[pid] | beta,sin,cos | zeros]
// Makes input_linear a plain dense-f16 GEMM.
// ---------------------------------------------------------------------------
__global__ void h0ext_k(const int* snz, const int* pid, const float* seq_emb,
                        const float* pos_emb, const float* temb, _Float16* h0e) {
  int idx = blockIdx.x * 256 + threadIdx.x;
  if (idx >= N_ * ILK) return;
  int n = idx / ILK, k = idx % ILK;
  float v = 0.f;
  if (k < HID)           v = seq_emb[(size_t)snz[n] * HID + k] + pos_emb[(size_t)pid[n] * HID + k];
  else if (k < HID + 3)  v = temb[(size_t)n * 4 + (k - HID)];
  h0e[idx] = (_Float16)v;
}

// ---------------------------------------------------------------------------
// WMMA GEMM: O = act(A[M,K] * W + bias).
// One wave computes NT 16x16 output tiles sharing one A fragment.
// B fragment: two 8-elem chunks from Wt[bn] at s, s+16, s = kt*32 + (lane>=16?8:0).
// A fragment: two 8-elem chunks at s, s+16 with s = kt*32 + (lane>=16?8:0).
// A-source modes: 0 dense f16 | 1 dense f32 | 3 edge-concat | 4 [h|aggm].
// Requires M % 16 == 0, Ntiles % NT == 0.
// ---------------------------------------------------------------------------
struct GemmP {
  int M, Ktiles, Ntiles, Kpad;
  int Nstore, ldo, act;              // act: 0 none, 1 silu, 2 relu
  const _Float16* Wt; const float* biasp;
  const float* Af; const _Float16* Ah; int lda;
  const float* hsrc; const float* radial; const float* rinv;
  const float* eemb; const int* ctx; const int* itr; int Ec;    // mode 3
  const float* aggm;                                            // mode 4
  float* Of; _Float16* Oh;
};

template <int AMODE, int NT>
__global__ __launch_bounds__(256) void gemm_k(GemmP p) {
  const int lane = threadIdx.x & 31;
  const int mtiles = p.M >> 4;                   // M % 16 == 0
  const int ngroups = p.Ntiles / NT;
  const int tile = blockIdx.x * 8 + (threadIdx.x >> 5);
  if (tile >= mtiles * ngroups) return;
  const int tm = tile % mtiles, tg = tile / mtiles;
  const int m0 = tm << 4;
  const int ar = m0 + (lane & 15);
  const int bn0 = tg * (16 * NT) + (lane & 15);
  const int bn1 = bn0 + 16;
  const int khiA = (lane >= 16) ? 8 : 0;
  const int khiB = (lane >= 16) ? 8 : 0;

  const _Float16* wr0 = p.Wt + (size_t)bn0 * p.Kpad + khiB;
  const _Float16* wr1 = p.Wt + (size_t)((NT == 2) ? bn1 : bn0) * p.Kpad + khiB;

  FragC c0, c1;
  {
    float bv0 = p.biasp[bn0];
#pragma unroll
    for (int r = 0; r < 8; ++r) c0.e[r] = bv0;
    if constexpr (NT == 2) {
      float bv1 = p.biasp[bn1];
#pragma unroll
      for (int r = 0; r < 8; ++r) c1.e[r] = bv1;
    }
  }

  auto mm = [&](const FragA& a, int k32) {
    FragA b0;
    b0.h8[0] = *(const v8h*)(wr0 + k32);
    b0.h8[1] = *(const v8h*)(wr0 + k32 + 16);
    c0.v = __builtin_amdgcn_wmma_f32_16x16x32_f16(false, a.v, false, b0.v, (short)0, c0.v,
                                                  false, false);
    if constexpr (NT == 2) {
      FragA b1;
      b1.h8[0] = *(const v8h*)(wr1 + k32);
      b1.h8[1] = *(const v8h*)(wr1 + k32 + 16);
      c1.v = __builtin_amdgcn_wmma_f32_16x16x32_f16(false, a.v, false, b1.v, (short)0, c1.v,
                                                    false, false);
    }
  };

  if constexpr (AMODE == 0) {
    const _Float16* ap = p.Ah + (size_t)ar * p.lda + khiA;
    for (int kt = 0; kt < p.Ktiles; ++kt) {
      FragA a;
      a.h8[0] = *(const v8h*)(ap + (kt << 5));
      a.h8[1] = *(const v8h*)(ap + (kt << 5) + 16);
      if (kt + 1 < p.Ktiles) __builtin_prefetch(ap + (kt << 5) + 32, 0, 1);
      mm(a, kt << 5);
    }
  } else if constexpr (AMODE == 1) {
    const float* ap = p.Af + (size_t)ar * p.lda + khiA;
    for (int kt = 0; kt < p.Ktiles; ++kt) {
      FragA a;
      a.h8[0] = cvt8(ap + (kt << 5));
      a.h8[1] = cvt8(ap + (kt << 5) + 16);
      mm(a, kt << 5);
    }
  } else if constexpr (AMODE == 3) {
    // Edge concat [h[row] | h[col] | radial*rinv | eattr | 0], K = 304/320.
    int e = ar, row, col, cls;
    if (e < p.Ec) { row = e / KNN; col = p.ctx[e]; cls = 0; }
    else { int e2 = e - p.Ec; row = e2 / KNN; col = p.itr[e2]; cls = 1; }
    const float* pr = p.hsrc + (size_t)row * HID + khiA;
    const float* pc = p.hsrc + (size_t)col * HID + khiA;
    const float* rd = p.radial + (size_t)e * 16 + khiA;
    const float* em = p.eemb + cls * 32 + khiA;
#pragma unroll
    for (int kt = 0; kt < 4; ++kt) {         // kt 0..3: h[row]
      FragA a;
      a.h8[0] = cvt8(pr + (kt << 5));
      a.h8[1] = cvt8(pr + (kt << 5) + 16);
      mm(a, kt << 5);
    }
#pragma unroll
    for (int kt = 0; kt < 4; ++kt) {         // kt 4..7: h[col]
      FragA a;
      a.h8[0] = cvt8(pc + (kt << 5));
      a.h8[1] = cvt8(pc + (kt << 5) + 16);
      mm(a, (kt + 4) << 5);
    }
    {                                        // kt 8: radial*rinv | eattr[0..15]
      FragA a;
      const float* rv = p.rinv + khiA;
      v8h r;
#pragma unroll
      for (int i = 0; i < 8; ++i) r[i] = (_Float16)(rd[i] * rv[i]);
      a.h8[0] = r;
      a.h8[1] = cvt8(em);
      mm(a, 8 << 5);
    }
    {                                        // kt 9: eattr[16..31] | zero
      FragA a;
      a.h8[0] = cvt8(em + 16);
      a.h8[1] = zero8();
      mm(a, 9 << 5);
    }
  } else {  // AMODE 4: [h | aggm], K = 256.
    const float* ph = p.Af + (size_t)ar * HID + khiA;
    const float* pa = p.aggm + (size_t)ar * HID + khiA;
#pragma unroll
    for (int kt = 0; kt < 4; ++kt) {
      FragA a;
      a.h8[0] = cvt8(ph + (kt << 5));
      a.h8[1] = cvt8(ph + (kt << 5) + 16);
      mm(a, kt << 5);
    }
#pragma unroll
    for (int kt = 0; kt < 4; ++kt) {
      FragA a;
      a.h8[0] = cvt8(pa + (kt << 5));
      a.h8[1] = cvt8(pa + (kt << 5) + 16);
      mm(a, (kt + 4) << 5);
    }
  }

#pragma unroll
  for (int r = 0; r < 8; ++r) {
    int row = m0 + r + ((lane >= 16) ? 8 : 0);      // always < M
    float v = c0.e[r];
    if (p.act == 1)      v = v / (1.f + expf(-v));
    else if (p.act == 2) v = fmaxf(v, 0.f);
    if (bn0 < p.Nstore) {
      if (p.Of) p.Of[(size_t)row * p.ldo + bn0] = v;
      else      p.Oh[(size_t)row * p.ldo + bn0] = (_Float16)v;
    }
    if constexpr (NT == 2) {
      float v1 = c1.e[r];
      if (p.act == 1)      v1 = v1 / (1.f + expf(-v1));
      else if (p.act == 2) v1 = fmaxf(v1, 0.f);
      if (bn1 < p.Nstore) {
        if (p.Of) p.Of[(size_t)row * p.ldo + bn1] = v1;
        else      p.Oh[(size_t)row * p.ldo + bn1] = (_Float16)v1;
      }
    }
  }
}

// ---------------------------------------------------------------------------
// Radial features + global sum-of-squares (for F.normalize(dim=0)).
// ---------------------------------------------------------------------------
__global__ __launch_bounds__(256) void radial_k(const float* x, const int* ctx, const int* itr,
                                                float* radial, float* rssq) {
  int e = blockIdx.x * 256 + threadIdx.x;
  __shared__ float ss[16];
  if (threadIdx.x < 16) ss[threadIdx.x] = 0.f;
  __syncthreads();
  if (e < E_) {
    int row, col;
    if (e < EC_) { row = e / KNN; col = ctx[e]; }
    else { int e2 = e - EC_; row = e2 / KNN; col = itr[e2]; }
    float cd[12];
    for (int j = 0; j < 12; ++j) cd[j] = x[(size_t)row * 12 + j] - x[(size_t)col * 12 + j];
    for (int c = 0; c < 4; ++c)
      for (int d = 0; d < 4; ++d) {
        float r = cd[c*3]*cd[d*3] + cd[c*3+1]*cd[d*3+1] + cd[c*3+2]*cd[d*3+2];
        radial[(size_t)e * 16 + c * 4 + d] = r;
        atomicAdd(&ss[c * 4 + d], r * r);
      }
  }
  __syncthreads();
  if (threadIdx.x < 16) atomicAdd(&rssq[threadIdx.x], ss[threadIdx.x]);
}

__global__ void rinv_k(const float* rssq, float* rinv) {
  int q = threadIdx.x;
  if (q < 16) rinv[q] = 1.f / fmaxf(sqrtf(rssq[q]), 1e-12f);
}

// ---------------------------------------------------------------------------
// Per-node gather aggregation (node n owns its 9 ctx + 9 inter edges).
// ---------------------------------------------------------------------------
__global__ __launch_bounds__(128) void nodeagg_k(const _Float16* m2, const float* x,
                                                 const float* wc, const int* ctx,
                                                 const int* itr, float* aggm, float* xn) {
  int n = blockIdx.x, tid = threadIdx.x;
  float s = 0.f;
  for (int t = 0; t < KNN; ++t) s += (float)m2[(size_t)(n * KNN + t) * HID + tid];
  for (int t = 0; t < KNN; ++t) s += (float)m2[(size_t)(EC_ + n * KNN + t) * HID + tid];
  aggm[(size_t)n * HID + tid] = s;
  if (tid < 12) {
    int c = tid / 3;
    float xr = x[(size_t)n * 12 + tid];
    float acc = 0.f;
    for (int t = 0; t < 2 * KNN; ++t) {
      int e   = (t < KNN) ? (n * KNN + t) : (EC_ + n * KNN + (t - KNN));
      int col = (t < KNN) ? ctx[n * KNN + t] : itr[n * KNN + (t - KNN)];
      float cdv = xr - x[(size_t)col * 12 + tid];
      acc += cdv * wc[(size_t)e * 4 + c];
    }
    xn[(size_t)n * 12 + tid] = xr + acc / 18.f;   // coords_agg='mean', cnt==18
  }
}

__global__ void add_k(float* h, const float* src, int n) {
  int i = blockIdx.x * 256 + threadIdx.x;
  if (i < n) h[i] += src[i];
}

__global__ void zero_k(float* p, int n) {
  int i = blockIdx.x * blockDim.x + threadIdx.x;
  if (i < n) p[i] = 0.f;
}

__global__ void softmax_k(float* cden) {
  int g = blockIdx.x * 256 + threadIdx.x;
  if (g >= N_) return;
  float* row = cden + (size_t)g * KCLS;
  float mx = row[0];
  for (int k = 1; k < KCLS; ++k) mx = fmaxf(mx, row[k]);
  float s = 0.f;
  for (int k = 0; k < KCLS; ++k) { float e = expf(row[k] - mx); row[k] = e; s += e; }
  for (int k = 0; k < KCLS; ++k) row[k] /= s;
}

// ---------------------------------------------------------------------------
// Losses: masked MSE on eps, masked posterior-KL on sequence.
// ---------------------------------------------------------------------------
__global__ __launch_bounds__(256) void loss_k(
    const float* xf, const float* pn, const float* eps, const int* seg,
    const int* S0, const int* snz, const int* tB, const float* alphas,
    const float* abar, const float* cden, const int* dsflag, float* acc)
{
  int g = blockIdx.x * 256 + threadIdx.x;
  int tid = threadIdx.x;
  float lp = 0.f, ls = 0.f, mgf = 0.f;
  if (g < N_) {
    bool mg = (seg[g] == 1);
    mgf = mg ? 1.f : 0.f;
    int b = g >> 8, tb = tB[b];
    int ds = *dsflag;
    float s12 = 0.f;
    for (int j = 0; j < 12; ++j) {
      float ep = xf[(size_t)g * 12 + j] - pn[(size_t)g * 12 + j];
      float et = ds ? eps[(size_t)g * 12 + j] : 0.f;
      float d = ep - et;
      s12 += d * d;
    }
    lp = mgf * (s12 * 0.25f);
    float a = alphas[tb], ab = abar[tb];
    int s0 = S0[g], sn = snz[g];
    float pt[KCLS], pd[KCLS], st = 0.f, sd = 0.f;
    for (int k = 0; k < KCLS; ++k) {
      float ct = (k == sn) ? 1.f : 0.f;
      float c0 = (k == s0) ? 1.f : 0.f;
      float f1 = a * ct + (1.f - a) / KCLS;
      pt[k] = f1 * (ab * c0 + (1.f - ab) / KCLS); st += pt[k];
      pd[k] = f1 * (ab * cden[(size_t)g * KCLS + k] + (1.f - ab) / KCLS); sd += pd[k];
    }
    float kl = 0.f;
    for (int k = 0; k < KCLS; ++k) {
      float p = pt[k] / (st + 1e-8f);
      float q = pd[k] / (sd + 1e-8f);
      if (p > 0.f) kl += p * (logf(p) - logf(q + 1e-8f));
    }
    ls = mgf * kl;
  }
  __shared__ float r0[256], r1[256], r2[256];
  r0[tid] = lp; r1[tid] = ls; r2[tid] = mgf;
  __syncthreads();
  for (int s2 = 128; s2 > 0; s2 >>= 1) {
    if (tid < s2) { r0[tid] += r0[tid+s2]; r1[tid] += r1[tid+s2]; r2[tid] += r2[tid+s2]; }
    __syncthreads();
  }
  if (tid == 0) {
    atomicAdd(&acc[0], r0[0]);
    atomicAdd(&acc[1], r1[0]);
    atomicAdd(&acc[2], r2[0]);
  }
}

__global__ void finalize_k(const float* acc, const int* ds, const int* dq, float* out) {
  if (threadIdx.x == 0 && blockIdx.x == 0) {
    float denom = acc[2] + 1e-8f;
    out[0] = (*ds) ? acc[0] / denom : 0.f;
    out[1] = (*dq) ? acc[1] / denom : 0.f;
  }
}

// ---------------------------------------------------------------------------
static inline int gemm_blocks(int M, int Ntiles, int NT) {
  return ((M / 16) * (Ntiles / NT) + 7) / 8;
}

extern "C" void kernel_launch(void* const* d_in, const int* in_sizes, int n_in,
                              void* d_out, int out_size, void* d_ws, size_t ws_size,
                              hipStream_t stream) {
  // ---- inputs (setup_inputs dict insertion order; params flattened in
  //      insertion order, each _lin = {w, b}) ----
  const int*           S0    = (const int*)d_in[0];
  const float*         X0    = (const float*)d_in[1];
  const int*           pid   = (const int*)d_in[2];
  const int*           seg   = (const int*)d_in[5];
  const unsigned char* amask = (const unsigned char*)d_in[6];   // jax bool = 1 byte
  const int*           tB    = (const int*)d_in[7];
  const float*         eps   = (const float*)d_in[8];
  const float*         gum   = (const float*)d_in[9];
  const int*           ds    = (const int*)d_in[10];
  const int*           dq    = (const int*)d_in[11];
  const float* seq_emb  = (const float*)d_in[12];               // [20,128]
  const float* pos_emb  = (const float*)d_in[13];               // [300,128]
  const float* edge_emb = (const float*)d_in[14];               // [2,32]
  const float* il_w     = (const float*)d_in[15];               // [131,128]
  const float* il_b     = (const float*)d_in[16];
  struct GclW { const float *e1w,*e1b,*e2w,*e2b,*n1w,*n1b,*n2w,*n2b,*c1w,*c1b,*c2w; } gwp[3];
  for (int l = 0; l < 3; ++l) {
    int base = 17 + 11 * l;
    gwp[l].e1w = (const float*)d_in[base + 0];
    gwp[l].e1b = (const float*)d_in[base + 1];
    gwp[l].e2w = (const float*)d_in[base + 2];
    gwp[l].e2b = (const float*)d_in[base + 3];
    gwp[l].n1w = (const float*)d_in[base + 4];
    gwp[l].n1b = (const float*)d_in[base + 5];
    gwp[l].n2w = (const float*)d_in[base + 6];
    gwp[l].n2b = (const float*)d_in[base + 7];
    gwp[l].c1w = (const float*)d_in[base + 8];
    gwp[l].c1b = (const float*)d_in[base + 9];
    gwp[l].c2w = (const float*)d_in[base + 10];
  }
  const float* s1w = (const float*)d_in[50];
  const float* s1b = (const float*)d_in[51];
  const float* s2w = (const float*)d_in[52];
  const float* s2b = (const float*)d_in[53];
  const float* s3w = (const float*)d_in[54];
  const float* s3b = (const float*)d_in[55];
  float* out = (float*)d_out;

  // ---- workspace carve-out (deterministic every call) ----
  char* wp = (char*)d_ws;
  auto alloc = [&](size_t bytes) -> void* {
    void* p = wp;
    wp += (bytes + 255) & ~(size_t)255;
    return p;
  };
  float* betas   = (float*)alloc(101 * sizeof(float));
  float* alphasv = (float*)alloc(101 * sizeof(float));
  float* abar    = (float*)alloc(101 * sizeof(float));
  float* centers = (float*)alloc(B_ * 4 * sizeof(float));
  float* temb    = (float*)alloc((size_t)N_ * 4 * sizeof(float));
  float* pn      = (float*)alloc((size_t)N_ * 12 * sizeof(float));
  float* xA      = (float*)alloc((size_t)N_ * 12 * sizeof(float));
  float* xB      = (float*)alloc((size_t)N_ * 12 * sizeof(float));
  int*   snz     = (int*)alloc((size_t)N_ * sizeof(int));
  int*   ctxc    = (int*)alloc((size_t)EC_ * sizeof(int));
  int*   itrc    = (int*)alloc((size_t)EC_ * sizeof(int));
  _Float16* h0e  = (_Float16*)alloc((size_t)N_ * ILK * sizeof(_Float16));
  float* h       = (float*)alloc((size_t)N_ * HID * sizeof(float));
  float* radial  = (float*)alloc((size_t)E_ * 16 * sizeof(float));
  float* rssq    = (float*)alloc(16 * sizeof(float));
  float* rinv    = (float*)alloc(16 * sizeof(float));
  _Float16* bufE1 = (_Float16*)alloc((size_t)E_ * HID * sizeof(_Float16));
  _Float16* bufE2 = (_Float16*)alloc((size_t)E_ * HID * sizeof(_Float16));
  float* wcoef   = (float*)alloc((size_t)E_ * 4 * sizeof(float));
  float* aggm    = (float*)alloc((size_t)N_ * HID * sizeof(float));
  _Float16* bufN1 = (_Float16*)alloc((size_t)N_ * HID * sizeof(_Float16));
  float* bufN2   = (float*)alloc((size_t)N_ * HID * sizeof(float));
  float* cden    = (float*)alloc((size_t)N_ * KCLS * sizeof(float));
  float* acc     = (float*)alloc(4 * sizeof(float));

  // ---- f16 transposed/padded weights ----
  struct WT { _Float16* wt; float* bp; int Kpad; };
  auto prep = [&](const float* W, const float* bias, int K, int Nreal,
                  int Kpad, int Npad) -> WT {
    _Float16* wt = (_Float16*)alloc((size_t)Npad * Kpad * sizeof(_Float16));
    float* bp = (float*)alloc((size_t)Npad * sizeof(float));
    int total = Npad * Kpad;
    wprep_k<<<(total + 255) / 256, 256, 0, stream>>>(W, bias, wt, bp, K, Nreal, Kpad, Npad);
    return WT{wt, bp, Kpad};
  };
  WT w_il = prep(il_w, il_b, HID + 3, HID, ILK, HID);
  WT w_e1[3], w_e2[3], w_c1[3], w_c2[3], w_n1[3], w_n2[3];
  for (int l = 0; l < 3; ++l) {
    w_e1[l] = prep(gwp[l].e1w, gwp[l].e1b, 304, HID, 320, HID);
    w_e2[l] = prep(gwp[l].e2w, gwp[l].e2b, HID, HID, HID, HID);
    w_c1[l] = prep(gwp[l].c1w, gwp[l].c1b, HID, HID, HID, HID);
    w_c2[l] = prep(gwp[l].c2w, nullptr,    HID, C_,  HID, 16);
    w_n1[l] = prep(gwp[l].n1w, gwp[l].n1b, 2 * HID, HID, 2 * HID, HID);
    w_n2[l] = prep(gwp[l].n2w, gwp[l].n2b, HID, HID, HID, HID);
  }
  WT w_s1 = prep(s1w, s1b, HID, HID, HID, HID);
  WT w_s2 = prep(s2w, s2b, HID, HID, HID, HID);
  WT w_s3 = prep(s3w, s3b, HID, KCLS, HID, 32);

  // ---- pipeline ----
  sched_k<<<1, 1, 0, stream>>>(betas, alphasv, abar);
  center_k<<<B_, 256, 0, stream>>>(X0, amask, seg, centers);
  noise_k<<<N_ / 256, 256, 0, stream>>>(X0, eps, gum, S0, seg, tB, ds, dq,
                                        betas, abar, centers, pn, xA, temb, snz);
  edges_k<<<N_, 256, 0, stream>>>(pn, amask, seg, ctxc, itrc);
  h0ext_k<<<(N_ * ILK + 255) / 256, 256, 0, stream>>>(snz, pid, seq_emb, pos_emb, temb, h0e);

  // input_linear: dense f16 [N,160] -> h (f32)
  {
    GemmP p{};
    p.M = N_; p.Ktiles = 5; p.Ntiles = 8; p.Kpad = w_il.Kpad;
    p.Nstore = HID; p.ldo = HID; p.act = 0;
    p.Wt = w_il.wt; p.biasp = w_il.bp; p.Ah = h0e; p.lda = ILK; p.Of = h;
    gemm_k<0, 2><<<gemm_blocks(p.M, p.Ntiles, 2), 256, 0, stream>>>(p);
  }

  float* xc = xA;
  float* xn = xB;
  for (int l = 0; l < 3; ++l) {
    zero_k<<<1, 32, 0, stream>>>(rssq, 16);
    radial_k<<<(E_ + 255) / 256, 256, 0, stream>>>(xc, ctxc, itrc, radial, rssq);
    rinv_k<<<1, 16, 0, stream>>>(rssq, rinv);

    // edge1: staged concat (K=304), silu -> bufE1 (f16)
    {
      GemmP p{};
      p.M = E_; p.Ktiles = 10; p.Ntiles = 8; p.Kpad = w_e1[l].Kpad;
      p.Nstore = HID; p.ldo = HID; p.act = 1;
      p.Wt = w_e1[l].wt; p.biasp = w_e1[l].bp;
      p.hsrc = h; p.radial = radial; p.rinv = rinv; p.eemb = edge_emb;
      p.ctx = ctxc; p.itr = itrc; p.Ec = EC_;
      p.Oh = bufE1;
      gemm_k<3, 2><<<gemm_blocks(p.M, p.Ntiles, 2), 256, 0, stream>>>(p);
    }
    // edge2: silu -> bufE2 (= m)
    {
      GemmP p{};
      p.M = E_; p.Ktiles = 4; p.Ntiles = 8; p.Kpad = w_e2[l].Kpad;
      p.Nstore = HID; p.ldo = HID; p.act = 1;
      p.Wt = w_e2[l].wt; p.biasp = w_e2[l].bp;
      p.Ah = bufE1; p.lda = HID; p.Oh = bufE2;
      gemm_k<0, 2><<<gemm_blocks(p.M, p.Ntiles, 2), 256, 0, stream>>>(p);
    }
    // coord1: silu -> bufE1 (reuse)
    {
      GemmP p{};
      p.M = E_; p.Ktiles = 4; p.Ntiles = 8; p.Kpad = w_c1[l].Kpad;
      p.Nstore = HID; p.ldo = HID; p.act = 1;
      p.Wt = w_c1[l].wt; p.biasp = w_c1[l].bp;
      p.Ah = bufE2; p.lda = HID; p.Oh = bufE1;
      gemm_k<0, 2><<<gemm_blocks(p.M, p.Ntiles, 2), 256, 0, stream>>>(p);
    }
    // coord2: 128->4 -> wcoef (f32)
    {
      GemmP p{};
      p.M = E_; p.Ktiles = 4; p.Ntiles = 1; p.Kpad = w_c2[l].Kpad;
      p.Nstore = 4; p.ldo = 4; p.act = 0;
      p.Wt = w_c2[l].wt; p.biasp = w_c2[l].bp;
      p.Ah = bufE1; p.lda = HID; p.Of = wcoef;
      gemm_k<0, 1><<<gemm_blocks(p.M, p.Ntiles, 1), 256, 0, stream>>>(p);
    }
    nodeagg_k<<<N_, 128, 0, stream>>>(bufE2, xc, wcoef, ctxc, itrc, aggm, xn);
    // node1: [h | aggm] (K=256), silu -> bufN1
    {
      GemmP p{};
      p.M = N_; p.Ktiles = 8; p.Ntiles = 8; p.Kpad = w_n1[l].Kpad;
      p.Nstore = HID; p.ldo = HID; p.act = 1;
      p.Wt = w_n1[l].wt; p.biasp = w_n1[l].bp;
      p.Af = h; p.aggm = aggm; p.Oh = bufN1;
      gemm_k<4, 2><<<gemm_blocks(p.M, p.Ntiles, 2), 256, 0, stream>>>(p);
    }
    // node2: -> bufN2 (f32), then h += bufN2
    {
      GemmP p{};
      p.M = N_; p.Ktiles = 4; p.Ntiles = 8; p.Kpad = w_n2[l].Kpad;
      p.Nstore = HID; p.ldo = HID; p.act = 0;
      p.Wt = w_n2[l].wt; p.biasp = w_n2[l].bp;
      p.Ah = bufN1; p.lda = HID; p.Of = bufN2;
      gemm_k<0, 2><<<gemm_blocks(p.M, p.Ntiles, 2), 256, 0, stream>>>(p);
    }
    add_k<<<(N_ * HID) / 256, 256, 0, stream>>>(h, bufN2, N_ * HID);
    std::swap(xc, xn);
  }

  // sequence head: relu(s1) -> relu(s2) -> s3 -> softmax
  {
    GemmP p{};
    p.M = N_; p.Ktiles = 4; p.Ntiles = 8; p.Kpad = w_s1.Kpad;
    p.Nstore = HID; p.ldo = HID; p.act = 2;
    p.Wt = w_s1.wt; p.biasp = w_s1.bp; p.Af = h; p.lda = HID; p.Oh = bufN1;
    gemm_k<1, 2><<<gemm_blocks(p.M, p.Ntiles, 2), 256, 0, stream>>>(p);
  }
  {
    GemmP p{};
    p.M = N_; p.Ktiles = 4; p.Ntiles = 8; p.Kpad = w_s2.Kpad;
    p.Nstore = HID; p.ldo = HID; p.act = 2;
    p.Wt = w_s2.wt; p.biasp = w_s2.bp; p.Ah = bufN1; p.lda = HID; p.Oh = bufE1;  // reuse
    gemm_k<0, 2><<<gemm_blocks(p.M, p.Ntiles, 2), 256, 0, stream>>>(p);
  }
  {
    GemmP p{};
    p.M = N_; p.Ktiles = 4; p.Ntiles = 2; p.Kpad = w_s3.Kpad;
    p.Nstore = KCLS; p.ldo = KCLS; p.act = 0;
    p.Wt = w_s3.wt; p.biasp = w_s3.bp; p.Ah = bufE1; p.lda = HID; p.Of = cden;
    gemm_k<0, 2><<<gemm_blocks(p.M, p.Ntiles, 2), 256, 0, stream>>>(p);
  }
  softmax_k<<<N_ / 256, 256, 0, stream>>>(cden);

  zero_k<<<1, 32, 0, stream>>>(acc, 4);
  loss_k<<<N_ / 256, 256, 0, stream>>>(xc, pn, eps, seg, S0, snz, tB,
                                       alphasv, abar, cden, ds, acc);
  finalize_k<<<1, 32, 0, stream>>>(acc, ds, dq, out);
}
